// SeasonalityBlock_57286273794224
// MI455X (gfx1250) — matrix operand order repaired and checked by reading, not compiled, not run on hardware
//
#include <hip/hip_runtime.h>
#include <stdint.h>

// ---------------------------------------------------------------------------
// N-BEATS SeasonalityBlock forward on gfx1250 (MI455X).
// All GEMMs: C[M,N] = act(A[M,K](bf16) * Bw[N,K](bf16)^T + bias)
// Tile staging uses CDNA5 async memory->LDS (global_load_async_to_lds_b128,
// ASYNCcnt), double-buffered; math is v_wmma_f32_16x16x32_bf16.
// ---------------------------------------------------------------------------

typedef __attribute__((ext_vector_type(16))) __bf16 bf16x16;
typedef __attribute__((ext_vector_type(8)))  float  f32x8;

#define TILE_M 128
#define TILE_N 128
#define TILE_K 32
#define LDS_STRIDE 40   // 32 bf16 payload + 8 pad -> 80B row stride,
                        // 20-bank stride: 16 distinct banks over 16 rows

__device__ __forceinline__ void store_elem(float* p, float v)  { *p = v; }
__device__ __forceinline__ void store_elem(__bf16* p, float v) { *p = (__bf16)v; }

// --- GEMM: C = act(A * Bw^T + bias), A/Bw bf16, M%128==0, N%128==0, K%32==0 -
template <typename TOUT>
__global__ __launch_bounds__(256)
void gemm_async_wmma(const __bf16* __restrict__ A, const __bf16* __restrict__ Bw,
                     const float* __restrict__ bias, TOUT* __restrict__ C,
                     int M, int N, int K, int relu)
{
  // [buf][mat][row][col] : 2 * 2 * 128 * 40 * 2B = 40 KB
  __shared__ alignas(16) __bf16 sm[2][2][TILE_M][LDS_STRIDE];

  const int tid  = threadIdx.x;
  const int lane = tid & 31;
  const int wave = tid >> 5;
  const int wm   = wave & 3;   // 4 waves along M (32 rows each)
  const int wn   = wave >> 2;  // 2 waves along N (64 cols each)
  const long mBase = (long)blockIdx.x * TILE_M;
  const long nBase = (long)blockIdx.y * TILE_N;

  const int mfrag = lane & 15;
  const int hfrag = lane >> 4;

  // Async staging lane map: 8 rows x 4 chunks(16B) per wave-instruction.
  const int srow = wave * 16 + (lane >> 2);  // +0 / +8 over j
  const int sch  = lane & 3;                 // 16B chunk within 64B row payload
  const uint32_t ldsBase = (uint32_t)(uintptr_t)&sm[0][0][0][0];

  // Loop-invariant staging addresses: both buffers, A and B tiles, j=0/1.
  uint32_t stA[2][2], stB[2][2];
  #pragma unroll
  for (int b = 0; b < 2; ++b) {
    #pragma unroll
    for (int j = 0; j < 2; ++j) {
      stA[b][j] = ldsBase +
          ((((uint32_t)b * 2 + 0) * TILE_M + srow + j * 8) * LDS_STRIDE + sch * 8) * 2;
      stB[b][j] = ldsBase +
          ((((uint32_t)b * 2 + 1) * TILE_M + srow + j * 8) * LDS_STRIDE + sch * 8) * 2;
    }
  }
  const __bf16* gA0 = A  + (mBase + srow) * (long)K + sch * 8;
  const __bf16* gB0 = Bw + (nBase + srow) * (long)K + sch * 8;
  const long rowK8 = 8L * K;

  // 4 async b128 loads per thread per stage (wave: 32 lanes x 16B = 512B each)
  auto stage = [&](int b, int k0) {
    #pragma unroll
    for (int j = 0; j < 2; ++j) {
      asm volatile("global_load_async_to_lds_b128 %0, %1, off"
                   :: "v"(stA[b][j]), "v"((uint64_t)(gA0 + j * rowK8 + k0))
                   : "memory");
      asm volatile("global_load_async_to_lds_b128 %0, %1, off"
                   :: "v"(stB[b][j]), "v"((uint64_t)(gB0 + j * rowK8 + k0))
                   : "memory");
    }
  };

  f32x8 acc[2][4] = {};

  // Fragment gathers per ISA 7.12.2 bf16 layouts + 8 WMMAs on buffer 'cur'.
  auto compute = [&](int cur) {
    bf16x16 af[2], bfr[4];
    union { bf16x16 v; unsigned u[8]; } t;
    #pragma unroll
    for (int mi = 0; mi < 2; ++mi) {
      const int row = wm * 32 + mi * 16 + mfrag;
      #pragma unroll
      for (int v = 0; v < 8; ++v) {
        // A 16x32: lanes0-15 V0..3->K0..7,V4..7->K16..23; lanes16-31 +8/+8
        const int kk = ((v < 4) ? 0 : 16) + hfrag * 8 + (v & 3) * 2;
        t.u[v] = *(const unsigned*)&sm[cur][0][row][kk];
      }
      af[mi] = t.v;
    }
    #pragma unroll
    for (int ni = 0; ni < 4; ++ni) {
      const int col = wn * 64 + ni * 16 + mfrag;
      #pragma unroll
      for (int v = 0; v < 8; ++v) {
        // B 32x16: lanes0-15 V0..3->K0..7,V4..7->K16..23; lanes16-31 +8/+8
        const int kk = ((v < 4) ? 0 : 16) + hfrag * 8 + (v & 3) * 2;
        t.u[v] = *(const unsigned*)&sm[cur][1][col][kk];
      }
      bfr[ni] = t.v;
    }
    #pragma unroll
    for (int mi = 0; mi < 2; ++mi)
      #pragma unroll
      for (int ni = 0; ni < 4; ++ni)
        acc[mi][ni] = __builtin_amdgcn_wmma_f32_16x16x32_bf16(
            false, af[mi], false, bfr[ni], (short)0, acc[mi][ni], false, false);
  };

  stage(0, 0);
  int buf = 0;
  for (int k0 = 0; k0 + TILE_K < K; k0 += TILE_K) {
    stage(buf ^ 1, k0 + TILE_K);
    asm volatile("s_wait_asynccnt 0x4" ::: "memory");  // 'buf' tile landed
    __syncthreads();
    compute(buf);
    __syncthreads();  // all waves done reading 'buf' before it is re-staged
    buf ^= 1;
  }
  asm volatile("s_wait_asynccnt 0x0" ::: "memory");    // last tile landed
  __syncthreads();
  compute(buf);

  // Epilogue: D layout — lane(n=l&15,h=l>>4), VGPR v -> row v+8h, col n.
  #pragma unroll
  for (int mi = 0; mi < 2; ++mi) {
    #pragma unroll
    for (int ni = 0; ni < 4; ++ni) {
      const long gn = nBase + wn * 64 + ni * 16 + mfrag;
      const float bv = bias ? bias[gn] : 0.0f;
      #pragma unroll
      for (int v = 0; v < 8; ++v) {
        const long gm = mBase + wm * 32 + mi * 16 + hfrag * 8 + v;
        float val = acc[mi][ni][v] + bv;
        if (relu) val = fmaxf(val, 0.0f);
        store_elem(C + gm * (long)N + gn, val);
      }
    }
  }
}

// --- f32 -> bf16 conversion (n even) ---------------------------------------
__global__ void cvt_f32_to_bf16(const float* __restrict__ s,
                                __bf16* __restrict__ d, long n) {
  long i = ((long)blockIdx.x * blockDim.x + threadIdx.x) * 2;
  if (i < n) {
    float2 f = *(const float2*)(s + i);
    union { unsigned u; __bf16 h[2]; } c;
    c.h[0] = (__bf16)f.x;
    c.h[1] = (__bf16)f.y;
    *(unsigned*)(d + i) = c.u;
  }
}

// --- seasonality bases (bf16), transposed into [N,K] weight layout ---------
// basisT[n][2i] = cos(2*pi*i*n/L), basisT[n][2i+1] = sin(2*pi*i*n/L)
__global__ void build_basis(__bf16* __restrict__ bTb, __bf16* __restrict__ bTf) {
  int idx = blockIdx.x * blockDim.x + threadIdx.x;
  if (idx < 512 * 128) {                       // backcast basis^T [512][128]
    int n = idx >> 7, k = idx & 127, i = k >> 1;
    int r = (i * n) & 511;                     // exact mod keeps arg in [0,2pi)
    float ang = 6.28318530717958647692f * (float)r * (1.0f / 512.0f);
    bTb[idx] = (__bf16)((k & 1) ? sinf(ang) : cosf(ang));
  } else {
    int j = idx - 512 * 128;
    if (j < 128 * 128) {                       // forecast basis^T [128][128]
      int n = j >> 7, k = j & 127, i = k >> 1;
      int r = (i * n) & 127;
      float ang = 6.28318530717958647692f * (float)r * (1.0f / 128.0f);
      bTf[j] = (__bf16)((k & 1) ? sinf(ang) : cosf(ang));
    }
  }
}

// ---------------------------------------------------------------------------
extern "C" void kernel_launch(void* const* d_in, const int* in_sizes, int n_in,
                              void* d_out, int out_size, void* d_ws, size_t ws_size,
                              hipStream_t stream) {
  const float* x  = (const float*)d_in[0];
  const float* W1 = (const float*)d_in[1];
  const float* b1 = (const float*)d_in[2];
  const float* W2 = (const float*)d_in[3];
  const float* b2 = (const float*)d_in[4];
  const float* W3 = (const float*)d_in[5];
  const float* b3 = (const float*)d_in[6];
  const float* W4 = (const float*)d_in[7];
  const float* b4 = (const float*)d_in[8];
  const float* Wt = (const float*)d_in[9];
  float* out = (float*)d_out;

  const long B = 16384, U = 512, T = 128, LB = 512, LF = 128;

  // Workspace layout (bf16 elements, ~57 MB total):
  uint16_t* w = (uint16_t*)d_ws;
  size_t o = 0;
  __bf16* xb  = (__bf16*)(w + o); o += (size_t)B * U;   // x in bf16
  __bf16* hA  = (__bf16*)(w + o); o += (size_t)B * U;   // hidden ping
  __bf16* hB  = (__bf16*)(w + o); o += (size_t)B * U;   // hidden pong
  __bf16* th  = (__bf16*)(w + o); o += (size_t)B * T;   // theta
  __bf16* w1b = (__bf16*)(w + o); o += (size_t)U * U;
  __bf16* w2b = (__bf16*)(w + o); o += (size_t)U * U;
  __bf16* w3b = (__bf16*)(w + o); o += (size_t)U * U;
  __bf16* w4b = (__bf16*)(w + o); o += (size_t)U * U;
  __bf16* wtb = (__bf16*)(w + o); o += (size_t)T * U;
  __bf16* bTb = (__bf16*)(w + o); o += (size_t)LB * T;
  __bf16* bTf = (__bf16*)(w + o); o += (size_t)LF * T;

  dim3 blk(256);
  auto cvt = [&](const float* s, __bf16* d, long n) {
    cvt_f32_to_bf16<<<(unsigned)((n / 2 + 255) / 256), blk, 0, stream>>>(s, d, n);
  };
  cvt(x,  xb,  B * U);
  cvt(W1, w1b, U * U);
  cvt(W2, w2b, U * U);
  cvt(W3, w3b, U * U);
  cvt(W4, w4b, U * U);
  cvt(Wt, wtb, T * U);
  build_basis<<<(512 * 128 + 128 * 128 + 255) / 256, blk, 0, stream>>>(bTb, bTf);

  dim3 gMain((unsigned)(B / TILE_M), (unsigned)(U / TILE_N));   // N = 512
  dim3 gThin((unsigned)(B / TILE_M), 1);                        // N = 128

  gemm_async_wmma<__bf16><<<gMain, blk, 0, stream>>>(xb, w1b, b1, hA, B, U, LB, 1);
  gemm_async_wmma<__bf16><<<gMain, blk, 0, stream>>>(hA, w2b, b2, hB, B, U, U, 1);
  gemm_async_wmma<__bf16><<<gMain, blk, 0, stream>>>(hB, w3b, b3, hA, B, U, U, 1);
  gemm_async_wmma<__bf16><<<gMain, blk, 0, stream>>>(hA, w4b, b4, hB, B, U, U, 1);
  gemm_async_wmma<__bf16><<<gThin, blk, 0, stream>>>(hB, wtb, nullptr, th, B, T, U, 0);
  gemm_async_wmma<float ><<<gMain, blk, 0, stream>>>(th, bTb, nullptr, out, B, LB, T, 0);
  gemm_async_wmma<float ><<<gThin, blk, 0, stream>>>(th, bTf, nullptr, out + (size_t)B * LB, B, LF, T, 0);
}
